// WeightGRU_21217138442546
// MI455X (gfx1250) — matrix operand order repaired and checked by reading, not compiled, not run on hardware
//
#include <hip/hip_runtime.h>

// ---------------------------------------------------------------------------
// Persistent bf16-WMMA GRU rollout for MI455X (gfx1250, wave32).
//   B=32, T=2048, D=512, H=1024.
//   Encoder folded into gates: ig = x @ W_comb^T + b_comb,
//     W_comb = W_ih @ W_enc  [3H, D],  b_comb = b_gru + W_ih @ b_enc.
//   64 persistent workgroups; each owns 16 h-columns + 8 dec d-columns,
//   weight slices resident in LDS (~285 KB/WGP) in WMMA fragment layout.
//   2 grid syncs per timestep through L2. x/h broadcasts use async-to-LDS
//   loads (ASYNCcnt) overlapped with WMMA work; outputs streamed NT.
// ---------------------------------------------------------------------------

typedef __attribute__((ext_vector_type(16))) __bf16 v16bf;
typedef __attribute__((ext_vector_type(8)))  __bf16 v8bf;
typedef __attribute__((ext_vector_type(8)))  float  v8f;
typedef int          vi4 __attribute__((ext_vector_type(4)));
typedef unsigned int vu4 __attribute__((ext_vector_type(4)));

#define NB   32      // batch
#define DD   512     // input dim
#define HH   1024    // hidden dim
#define G3   3072    // 3*H
#define NWG  64
#define TPB  256

// per-WG swizzled weight slice: (3 gates * 48 chunks + 32 dec chunks) tiles,
// each tile = 32 lanes * 16 bf16 = 512 ushorts = 1024 bytes.
#define TILES_PER_WG   (3 * 48 + 32)          // 176
#define PER_WG_US      (TILES_PER_WG * 512)   // 90112 ushorts
#define PER_WG_BYTES   (TILES_PER_WG * 1024)  // 180224 bytes

// workspace layout (bytes)
#define WSW_OFF    0
#define BCOMB_OFF  (NWG * PER_WG_BYTES)            // 11,534,336
#define HBUF_OFF   (BCOMB_OFF + G3 * 4)            // +12,288
#define XBUF_OFF   (HBUF_OFF + NB * HH * 2)        // +65,536
#define BAR_OFF    (XBUF_OFF + NB * DD * 2)        // +32,768

// LDS layout (bytes)
#define L_WGATE  0                       // 3*48*1024 = 147456
#define L_WDEC   147456                  // 32*1024   =  32768
#define L_H      180224                  // 32*1024*2 =  65536  (bf16 h)
#define L_X      245760                  // 32*512*2  =  32768  (bf16 x)
#define L_R      278528                  // 32*16*4   =   2048
#define L_Z      280576                  //              2048
#define L_OUT    282624                  //              2048
#define L_DP     284672                  // 3*2*8*32*4 =  6144
#define L_EPS    290816                  // 32*8*4    =   1024
#define LDS_TOTAL 291840                 // 285 KB  (<= 320 KB/WGP)

#if __has_builtin(__builtin_amdgcn_global_load_async_to_lds_b128) && \
    __has_builtin(__builtin_amdgcn_s_wait_asynccnt)
#define HAVE_ASYNC 1
#else
#define HAVE_ASYNC 0
#endif

__device__ __forceinline__ void async_cp16(char* lds_dst, const char* gsrc) {
#if HAVE_ASYNC
    __builtin_amdgcn_global_load_async_to_lds_b128(
        (__attribute__((address_space(1))) vi4*)gsrc,
        (__attribute__((address_space(3))) vi4*)lds_dst, 0, 0);
#else
    *(vu4*)lds_dst = *(const vu4*)gsrc;
#endif
}
__device__ __forceinline__ void wait_async0() {
#if HAVE_ASYNC
    __builtin_amdgcn_s_wait_asynccnt(0);
#endif
}

__device__ __forceinline__ unsigned short f2bf(float f) {
    unsigned u = __builtin_bit_cast(unsigned, f);
    unsigned r = (u + 0x7FFFu + ((u >> 16) & 1u)) >> 16;
    return (unsigned short)r;
}
__device__ __forceinline__ float bf2f(unsigned short h) {
    unsigned u = ((unsigned)h) << 16;
    return __builtin_bit_cast(float, u);
}

// A-fragment (16x32 bf16, MxK): lanes 0-15 hold K {0..15}
// of row M=lane; lanes 16-31 hold K {16..31} of row M=lane-16.
__device__ __forceinline__ v16bf load_af(const unsigned short* act, int ldk,
                                         int mtile, int c, int lane) {
    int row = (mtile << 4) + (lane & 15);
    int kb  = (c << 5) + ((lane >> 4) << 4);
    const __bf16* p = (const __bf16*)(act + row * ldk + kb);
    v8bf lo = *(const v8bf*)(p);
    v8bf hi = *(const v8bf*)(p + 8);
    return __builtin_shufflevector(lo, hi, 0,1,2,3,4,5,6,7,8,9,10,11,12,13,14,15);
}

// ---------------------------------------------------------------------------
// Setup kernel 1: W_comb = W_ih @ W_enc -> swizzled bf16 fragments; b_comb.
// grid (4, 3072) x 128
__global__ void gru_prep_wcomb(const float* __restrict__ Wih,
                               const float* __restrict__ Wenc,
                               const float* __restrict__ benc,
                               const float* __restrict__ bgru,
                               unsigned short* __restrict__ wsw,
                               float* __restrict__ bcomb) {
    int gr = blockIdx.y;
    int d  = blockIdx.x * 128 + threadIdx.x;
    bool dob = (d == 0);
    float acc = 0.f, bias = 0.f;
    for (int hh = 0; hh < HH; ++hh) {
        float a = Wih[(size_t)gr * HH + hh];
        acc += a * Wenc[(size_t)hh * DD + d];
        if (dob) bias += a * benc[hh];
    }
    int g = (gr & (HH - 1)) >> 4, gate = gr >> 10, jl = gr & 15;
    int c = d >> 5, khalf = (d >> 4) & 1, i = d & 15;
    int ln = jl | (khalf << 4);
    wsw[(size_t)g * PER_WG_US + ((gate * 48 + c) * 512) + ln * 16 + i] = f2bf(acc);
    if (dob) bcomb[gr] = bgru[gr] + bias;
}

// Setup kernel 2: swizzle-convert W_hh (chunks 16..47) and W_dec (dec tiles).
// grid 16384 x 256
__global__ void gru_prep_conv(const float* __restrict__ Whh,
                              const float* __restrict__ Wdec,
                              unsigned short* __restrict__ wsw) {
    int idx = blockIdx.x * 256 + threadIdx.x;
    if (idx < G3 * HH) {                         // W_hh element
        int gr = idx >> 10, k = idx & (HH - 1);
        float v = Whh[idx];
        int g = (gr & (HH - 1)) >> 4, gate = gr >> 10, jl = gr & 15;
        int c = 16 + (k >> 5), khalf = (k >> 4) & 1, i = k & 15;
        int ln = jl | (khalf << 4);
        wsw[(size_t)g * PER_WG_US + ((gate * 48 + c) * 512) + ln * 16 + i] = f2bf(v);
    } else {                                     // W_dec element
        int e = idx - G3 * HH;                   // e < 1024*1024
        int r = e >> 10, k = e & (HH - 1);
        float v = Wdec[e];
        int g, n;
        if (r < DD) { g = r >> 3; n = r & 7; }
        else        { int rr = r - DD; g = rr >> 3; n = 8 + (rr & 7); }
        int c = k >> 5, khalf = (k >> 4) & 1, i = k & 15;
        int ln = n | (khalf << 4);
        wsw[(size_t)g * PER_WG_US + ((144 + c) * 512) + ln * 16 + i] = f2bf(v);
    }
}

// Setup kernel 3: h0 = 0, xbuf = bf16(x0), reset grid barrier.  128 x 256
__global__ void gru_prep_state(const float* __restrict__ x0,
                               unsigned short* __restrict__ hbuf,
                               unsigned short* __restrict__ xbuf,
                               unsigned* __restrict__ bar) {
    int idx = blockIdx.x * 256 + threadIdx.x;
    if (idx < NB * HH) hbuf[idx] = 0;
    if (idx < NB * DD) xbuf[idx] = f2bf(x0[idx]);
    if (idx == 0) { bar[0] = 0u; bar[1] = 0u; }
}

// ---------------------------------------------------------------------------
__device__ __forceinline__ void grid_sync(unsigned* bar, unsigned* localgen) {
    __threadfence();
    __syncthreads();
    if (threadIdx.x == 0) {
        unsigned g0 = *localgen;
        if (atomicAdd(&bar[0], 1u) == NWG - 1u) {
            bar[0] = 0u;
            __threadfence();
            atomicAdd(&bar[1], 1u);
        } else {
            while (__hip_atomic_load(&bar[1], __ATOMIC_ACQUIRE,
                                     __HIP_MEMORY_SCOPE_AGENT) == g0)
                __builtin_amdgcn_s_sleep(1);
        }
        *localgen = g0 + 1u;
    }
    __syncthreads();
}

// ---------------------------------------------------------------------------
// Persistent GRU kernel: grid 64 x 256, 291840 B dynamic LDS.
__global__ void gru_persistent(const unsigned short* __restrict__ wsw,
                               const float* __restrict__ bcomb,
                               unsigned short* __restrict__ hbuf,
                               unsigned short* __restrict__ xbuf,
                               unsigned* __restrict__ bar,
                               const float* __restrict__ eps,
                               const float* __restrict__ bn_,
                               const float* __restrict__ bdec,
                               float* __restrict__ out,
                               const int* __restrict__ steps, int Tmax) {
    extern __shared__ char smem[];
    const int tid  = threadIdx.x;
    const int lane = tid & 31;
    const int wid  = tid >> 5;
    const int g    = blockIdx.x;
    const int jl   = lane & 15;
    const int wgj0 = g * 16;    // first owned h-column
    const int wgd0 = g * 8;     // first owned dec d-column

    int T = steps[0]; if (T > Tmax) T = Tmax; if (T < 0) T = 0;
    unsigned localgen = 0u;
    const size_t TBD = (size_t)T * NB * DD;

    // ---- load this WG's weight slice into LDS (176 KB, b128 coalesced) ----
    {
        const vu4* src = (const vu4*)(wsw + (size_t)g * PER_WG_US);
        vu4* dst = (vu4*)(smem + L_WGATE);
        for (int i = tid; i < PER_WG_BYTES / 16; i += TPB) dst[i] = src[i];
    }
    // ---- stage initial h (x is broadcast at loop top every step) ----
    {
        const vu4* hs = (const vu4*)hbuf;
        vu4* hd = (vu4*)(smem + L_H);
        for (int i = tid; i < (NB * HH * 2) / 16; i += TPB) hd[i] = hs[i];
    }
    __syncthreads();

    // per-lane constants
    const int   gate  = wid >> 1;          // 0=r 1=z 2=n (valid for wid<6)
    const int   mtile = wid & 1;
    float bgate = 0.f;
    if (wid < 6) bgate = bcomb[gate * HH + wgj0 + jl];
    const float bnv   = bn_[wgj0 + jl];
    const float bdv   = bdec[(jl < 8) ? (wgd0 + jl) : (DD + wgd0 + (jl - 8))];

    const unsigned short* xlds = (const unsigned short*)(smem + L_X);
    const unsigned short* hlds = (const unsigned short*)(smem + L_H);
    float* rlds = (float*)(smem + L_R);
    float* zlds = (float*)(smem + L_Z);
    float* olds = (float*)(smem + L_OUT);
    float* dpls = (float*)(smem + L_DP);
    const float* epsl = (const float*)(smem + L_EPS);

    const int XV16 = (NB * DD * 2) / 16;   // 2048 x-fragments of 16B
    const int HV16 = (NB * HH * 2) / 16;   // 4096 h-fragments of 16B

    for (int t = 0; t < T; ++t) {
        // ---- (A) issue async broadcast of x_t into LDS (hidden by h-WMMAs)
        for (int i = tid; i < XV16; i += TPB)
            async_cp16(smem + L_X + i * 16, (const char*)xbuf + i * 16);

        // ---- (B) idle waves prefetch this step's eps slice (NT stream) ----
        if (wid >= 6) {
            int lcl = tid - 192;           // 0..63
            int m = lcl >> 1, half = lcl & 1;
            const vu4* src = (const vu4*)(eps + (size_t)t * (NB * DD)
                                          + (size_t)m * DD + wgd0 + half * 4);
            vu4 v = __builtin_nontemporal_load(src);
            *(vu4*)(smem + L_EPS + m * 32 + half * 16) = v;
        }

        // ---- (C) gates, h-part: hg = h @ W_hh^T (h already in LDS) ----
        v8f accx = {}; v8f acch = {};
        if (wid < 6) {
            const char* bb = smem + L_WGATE + gate * (48 * 1024);
            for (int c = 0; c < 32; ++c) {
                v16bf a = load_af(hlds, HH, mtile, c, lane);
                v16bf b = *(const v16bf*)(bb + (16 + c) * 1024 + lane * 32);
                acch = __builtin_amdgcn_wmma_f32_16x16x32_bf16(
                           false, a, false, b, (short)0, acch, false, false);
            }
        }
        // ---- (D) x_t has landed in LDS ----
        wait_async0();
        __syncthreads();

        // ---- (E) gates, x-part: ig = x @ W_comb^T ----
        if (wid < 6) {
            const char* bb = smem + L_WGATE + gate * (48 * 1024);
            for (int c = 0; c < 16; ++c) {
                v16bf a = load_af(xlds, DD, mtile, c, lane);
                v16bf b = *(const v16bf*)(bb + c * 1024 + lane * 32);
                accx = __builtin_amdgcn_wmma_f32_16x16x32_bf16(
                           false, a, false, b, (short)0, accx, false, false);
            }
            if (gate < 2) {
                float* dst = (gate == 0) ? rlds : zlds;
                for (int p = 0; p < 8; ++p) {
                    int M = mtile * 16 + p + ((lane >> 4) << 3);
                    float v = accx[p] + acch[p] + bgate;
                    dst[M * 16 + jl] = 1.f / (1.f + __expf(-v));
                }
            }
        }
        __syncthreads();

        // ---- (F) n-gate waves combine, h_new slice -> global hbuf ----
        if (wid >= 4 && wid < 6) {
            for (int p = 0; p < 8; ++p) {
                int M = mtile * 16 + p + ((lane >> 4) << 3);
                float r  = rlds[M * 16 + jl];
                float z  = zlds[M * 16 + jl];
                float nn = tanhf(accx[p] + bgate + r * (acch[p] + bnv));
                float ho = bf2f(hlds[M * HH + wgj0 + jl]);
                float hn = nn + z * (ho - nn);
                hbuf[M * HH + wgj0 + jl] = f2bf(hn);
            }
        }
        grid_sync(bar, &localgen);          // h_new visible everywhere

        // ---- (G) async reload of full h_new into LDS ----
        for (int i = tid; i < HV16; i += TPB)
            async_cp16(smem + L_H + i * 16, (const char*)hbuf + i * 16);
        wait_async0();
        __syncthreads();

        // ---- (H) decoder: h_new @ W_dec^T, K split over 4 wave-groups ----
        v8f dacc = {};
        {
            int kg = wid >> 1, m = wid & 1;
            const char* db = smem + L_WDEC;
            for (int cc = 0; cc < 8; ++cc) {
                int c = kg * 8 + cc;
                v16bf a = load_af(hlds, HH, m, c, lane);
                v16bf b = *(const v16bf*)(db + c * 1024 + lane * 32);
                dacc = __builtin_amdgcn_wmma_f32_16x16x32_bf16(
                           false, a, false, b, (short)0, dacc, false, false);
            }
            if (kg > 0) {
                float* pp = dpls + (((kg - 1) * 2 + m) * 8) * 32;
                for (int p = 0; p < 8; ++p) pp[p * 32 + lane] = dacc[p];
            }
        }
        __syncthreads();
        if (wid < 2) {                       // kg==0 waves reduce + bias
            int m = wid;
            for (int p = 0; p < 8; ++p) {
                float v = dacc[p] + bdv;
                for (int q = 0; q < 3; ++q)
                    v += dpls[((q * 2 + m) * 8 + p) * 32 + lane];
                int M = m * 16 + p + ((lane >> 4) << 3);
                olds[M * 16 + jl] = v;
            }
        }
        __syncthreads();

        // ---- (I) elementwise sample; NT-stream outputs; stage x_next ----
        {
            int m  = tid >> 3;               // 0..31
            int dl = tid & 7;                // 0..7
            float mu = olds[m * 16 + dl];
            float ls = olds[m * 16 + 8 + dl];
            float sg = __expf(ls);
            int   dg = wgd0 + dl;
            size_t eo = (size_t)t * (NB * DD) + (size_t)m * DD + dg;
            float xn = mu + sg * epsl[m * 8 + dl];
            __builtin_nontemporal_store(xn, out + eo);            // samples
            __builtin_nontemporal_store(mu, out + TBD + eo);      // means
            __builtin_nontemporal_store(sg, out + 2 * TBD + eo);  // sigmas
            xbuf[m * DD + dg] = f2bf(xn);
        }
        grid_sync(bar, &localgen);          // x_next visible everywhere
    }
}

// ---------------------------------------------------------------------------
extern "C" void kernel_launch(void* const* d_in, const int* in_sizes, int n_in,
                              void* d_out, int out_size, void* d_ws, size_t ws_size,
                              hipStream_t stream) {
    const float* x0   = (const float*)d_in[0];
    const float* eps  = (const float*)d_in[1];
    const float* Wenc = (const float*)d_in[2];
    const float* benc = (const float*)d_in[3];
    const float* Wih  = (const float*)d_in[4];
    const float* Whh  = (const float*)d_in[5];
    const float* bgru = (const float*)d_in[6];
    const float* bn   = (const float*)d_in[7];
    const float* Wdec = (const float*)d_in[8];
    const float* bdec = (const float*)d_in[9];
    const int*   steps = (const int*)d_in[10];
    float* out = (float*)d_out;

    int T = in_sizes[1] / (NB * DD);

    char* ws = (char*)d_ws;
    unsigned short* wsw   = (unsigned short*)(ws + WSW_OFF);
    float*          bcomb = (float*)(ws + BCOMB_OFF);
    unsigned short* hbuf  = (unsigned short*)(ws + HBUF_OFF);
    unsigned short* xbuf  = (unsigned short*)(ws + XBUF_OFF);
    unsigned*       bar   = (unsigned*)(ws + BAR_OFF);

    gru_prep_wcomb<<<dim3(DD / 128, G3), 128, 0, stream>>>(Wih, Wenc, benc,
                                                           bgru, wsw, bcomb);
    gru_prep_conv<<<(G3 * HH + DD * 2 * HH) / 256, 256, 0, stream>>>(Whh, Wdec, wsw);
    gru_prep_state<<<128, 256, 0, stream>>>(x0, hbuf, xbuf, bar);

    (void)hipFuncSetAttribute((const void*)gru_persistent,
                              hipFuncAttributeMaxDynamicSharedMemorySize, LDS_TOTAL);
    gru_persistent<<<NWG, TPB, LDS_TOTAL, stream>>>(wsw, bcomb, hbuf, xbuf, bar,
                                                    eps, bn, bdec, out, steps, T);
}
